// _PaddedAttn_15358803051121
// MI455X (gfx1250) — matrix operand order repaired and checked by reading, not compiled, not run on hardware
//
#include <hip/hip_runtime.h>
#include <math.h>

// ---------------------------------------------------------------------------
// ViT attention block for MI455X (gfx1250, wave32, WMMA f32_16x16x32_f16).
// B=16, N=1024, C=768, H=12, D=64.
// Pipeline: f32->f16 convert, QKV GEMM (WMMA), flash attention (WMMA +
// online softmax), projection GEMM (WMMA).
// ---------------------------------------------------------------------------

typedef __attribute__((ext_vector_type(16))) _Float16 v16h;
typedef __attribute__((ext_vector_type(8)))  float    v8f;

union Frag {
    v16h h;
    uint4 u[2];
};

#define BATCH 16
#define SEQ   1024
#define CDIM  768
#define HEADS 12
#define HDIM  64
#define QKVN  (3 * CDIM)   // 2304

// -------------------------------- convert ----------------------------------
__global__ void __launch_bounds__(256)
attn_cvt_f16(const float* __restrict__ src, _Float16* __restrict__ dst, int n) {
    int i = blockIdx.x * blockDim.x + threadIdx.x;
    if (i < n) dst[i] = (_Float16)src[i];
}

// ------------------------- shared WMMA GEMM mainloop ------------------------
// One wave computes a 32(M) x 64(N) tile of A(MxK) @ W(NxK)^T, K=768.
// A, W row-major f16 with K contiguous -> all fragment loads are contiguous.
__device__ __forceinline__ void wmma_tile_32x64(
        const _Float16* __restrict__ A, const _Float16* __restrict__ W,
        int lda, int ldw, int K, int mbase, int nbase, int lane,
        v8f acc[2][4]) {
    const int ln = lane & 15, lh = lane >> 4;
    const _Float16* arow0 = A + (size_t)(mbase + ln) * lda;
    const _Float16* arow1 = A + (size_t)(mbase + 16 + ln) * lda;
    const _Float16* brow0 = W + (size_t)(nbase + 0 * 16 + ln) * ldw;
    const _Float16* brow1 = W + (size_t)(nbase + 1 * 16 + ln) * ldw;
    const _Float16* brow2 = W + (size_t)(nbase + 2 * 16 + ln) * ldw;
    const _Float16* brow3 = W + (size_t)(nbase + 3 * 16 + ln) * ldw;
    const _Float16* brows[4] = {brow0, brow1, brow2, brow3};

    for (int k0 = 0; k0 < K; k0 += 32) {
        // A fragments: 16x32, lane = row, two contiguous 16B chunks along K.
        Frag a0, a1;
        a0.u[0] = *(const uint4*)(arow0 + k0 + 8 * lh);
        a0.u[1] = *(const uint4*)(arow0 + k0 + 16 + 8 * lh);
        a1.u[0] = *(const uint4*)(arow1 + k0 + 8 * lh);
        a1.u[1] = *(const uint4*)(arow1 + k0 + 16 + 8 * lh);
        if (k0 + 32 < K) {
            __builtin_prefetch(arow0 + k0 + 32, 0, 3);
            __builtin_prefetch(arow1 + k0 + 32, 0, 3);
        }
#pragma unroll
        for (int nt = 0; nt < 4; ++nt) {
            // B fragment: 32x16, lane = col, two contiguous 16B chunks along K.
            const _Float16* bp = brows[nt] + k0 + 8 * lh;
            Frag b;
            b.u[0] = *(const uint4*)(bp);
            b.u[1] = *(const uint4*)(bp + 16);
            acc[0][nt] = __builtin_amdgcn_wmma_f32_16x16x32_f16(
                false, a0.h, false, b.h, (short)0, acc[0][nt], false, false);
            acc[1][nt] = __builtin_amdgcn_wmma_f32_16x16x32_f16(
                false, a1.h, false, b.h, (short)0, acc[1][nt], false, false);
        }
    }
}

// ------------------------------- QKV GEMM ----------------------------------
// qkv = x @ qkv_w^T + qkv_b, scattered into q (scaled by D^-0.5), k, vT (f16).
__global__ void __launch_bounds__(128)
attn_qkv_gemm(const _Float16* __restrict__ x16, const _Float16* __restrict__ w16,
              const float* __restrict__ bias,
              _Float16* __restrict__ q16, _Float16* __restrict__ k16,
              _Float16* __restrict__ vT16) {
    const int wave = (int)((blockIdx.x * blockDim.x + threadIdx.x) >> 5);
    const int lane = threadIdx.x & 31;
    const int NT = QKVN / 64;                 // 36 n-tiles
    const int mt = wave / NT, nt = wave % NT;
    const int mbase = mt * 32, nbase = nt * 64;
    const int ln = lane & 15, lh = lane >> 4;

    v8f acc[2][4];
    const v8f vzero = {0.f, 0.f, 0.f, 0.f, 0.f, 0.f, 0.f, 0.f};
#pragma unroll
    for (int i = 0; i < 2; ++i)
#pragma unroll
        for (int j = 0; j < 4; ++j) acc[i][j] = vzero;

    wmma_tile_32x64(x16, w16, CDIM, CDIM, CDIM, mbase, nbase, lane, acc);

#pragma unroll
    for (int t = 0; t < 4; ++t) {
        const int j = nbase + t * 16 + ln;    // output column in [0, 2304)
        const float bj = bias[j];
        const int which = j / CDIM;           // 0=q 1=k 2=v (uniform per tile)
        const int c = j % CDIM;
        const int h = c >> 6, d = c & 63;
#pragma unroll
        for (int mi = 0; mi < 2; ++mi) {
#pragma unroll
            for (int r = 0; r < 8; ++r) {
                const int i = mbase + mi * 16 + r + 8 * lh;   // token row
                const int bb = i >> 10, ntok = i & 1023;
                const float val = acc[mi][t][r] + bj;
                const size_t bh = (size_t)(bb * HEADS + h);
                if (which == 0) {
                    q16[(bh * SEQ + ntok) * HDIM + d] = (_Float16)(val * 0.125f);
                } else if (which == 1) {
                    k16[(bh * SEQ + ntok) * HDIM + d] = (_Float16)val;
                } else {
                    vT16[(bh * HDIM + d) * SEQ + ntok] = (_Float16)val;
                }
            }
        }
    }
}

// --------------------------- flash attention core ---------------------------
// One wave: 16 query rows of one (b,h); online softmax over 1024 keys in
// 32-key chunks. 8 WMMA per chunk (4 score, 4 P@V).
__global__ void __launch_bounds__(128)
attn_flash(const _Float16* __restrict__ q16, const _Float16* __restrict__ k16,
           const _Float16* __restrict__ vT16, const float* __restrict__ mask,
           _Float16* __restrict__ o16) {
    __shared__ _Float16 plds[4][16 * 32];     // 1KB P-staging per wave

    const int wid = threadIdx.x >> 5;
    const int lane = threadIdx.x & 31;
    const int gw = (int)(blockIdx.x * 4 + wid);
    const int qt = gw & 63;                   // query tile within head
    const int bh = gw >> 6;                   // (b*HEADS + h)
    const int bb = bh / HEADS;
    const int h = bh % HEADS;
    const int ln = lane & 15, lh = lane >> 4;
    const int qbase = qt * 16;

    // Q fragments for K-dim = 64 (two 16x32 A-frags), contiguous loads.
    const _Float16* qrow = q16 + ((size_t)bh * SEQ + qbase + ln) * HDIM;
    Frag aq0, aq1;
    aq0.u[0] = *(const uint4*)(qrow + 16 * lh);
    aq0.u[1] = *(const uint4*)(qrow + 16 * lh + 8);
    aq1.u[0] = *(const uint4*)(qrow + 32 + 16 * lh);
    aq1.u[1] = *(const uint4*)(qrow + 32 + 16 * lh + 8);

    const v8f vzero = {0.f, 0.f, 0.f, 0.f, 0.f, 0.f, 0.f, 0.f};
    v8f acc[4];
#pragma unroll
    for (int dt = 0; dt < 4; ++dt) acc[dt] = vzero;
    float mrow[8], ssum[8];
#pragma unroll
    for (int r = 0; r < 8; ++r) { mrow[r] = -INFINITY; ssum[r] = 0.f; }

    const float* maskb = mask + (size_t)bb * SEQ * SEQ;
    const _Float16* kbh = k16 + (size_t)bh * SEQ * HDIM;
    const _Float16* vbh = vT16 + (size_t)bh * HDIM * SEQ;
    _Float16* pl = plds[wid];

    for (int kc = 0; kc < 32; ++kc) {
        const int keybase = kc * 32;
        v8f s[2];
#pragma unroll
        for (int t = 0; t < 2; ++t) {
            const _Float16* krow = kbh + (size_t)(keybase + t * 16 + ln) * HDIM;
            Frag bk0, bk1;
            bk0.u[0] = *(const uint4*)(krow + 16 * lh);
            bk0.u[1] = *(const uint4*)(krow + 16 * lh + 8);
            bk1.u[0] = *(const uint4*)(krow + 32 + 16 * lh);
            bk1.u[1] = *(const uint4*)(krow + 32 + 16 * lh + 8);
            s[t] = __builtin_amdgcn_wmma_f32_16x16x32_f16(
                false, aq0.h, false, bk0.h, (short)0, vzero, false, false);
            s[t] = __builtin_amdgcn_wmma_f32_16x16x32_f16(
                false, aq1.h, false, bk1.h, (short)0, s[t], false, false);
        }
        // additive mask: element (m = r + 8*lh, n = ln) of each 16-wide tile
#pragma unroll
        for (int t = 0; t < 2; ++t) {
            const float* mp = maskb + (size_t)(qbase + 8 * lh) * SEQ +
                              keybase + t * 16 + ln;
#pragma unroll
            for (int r = 0; r < 8; ++r) s[t][r] += mp[(size_t)r * SEQ];
        }
        // online softmax (reductions across the 16 lanes of each half)
#pragma unroll
        for (int r = 0; r < 8; ++r) {
            float cm = fmaxf(s[0][r], s[1][r]);
#pragma unroll
            for (int off = 1; off < 16; off <<= 1)
                cm = fmaxf(cm, __shfl_xor(cm, off, 32));
            const float nm = fmaxf(mrow[r], cm);
            const float corr = __expf(mrow[r] - nm);
            mrow[r] = nm;
            const float p0 = __expf(s[0][r] - nm);
            const float p1 = __expf(s[1][r] - nm);
            s[0][r] = p0; s[1][r] = p1;
            float rs = p0 + p1;
#pragma unroll
            for (int off = 1; off < 16; off <<= 1)
                rs += __shfl_xor(rs, off, 32);
            ssum[r] = ssum[r] * corr + rs;
#pragma unroll
            for (int dt = 0; dt < 4; ++dt) acc[dt][r] *= corr;
        }
        // stage P (C-layout) in LDS, reload as 16x32 A-fragment
#pragma unroll
        for (int t = 0; t < 2; ++t)
#pragma unroll
            for (int r = 0; r < 8; ++r)
                pl[(r + 8 * lh) * 32 + t * 16 + ln] = (_Float16)s[t][r];
        asm volatile("s_wait_dscnt 0" ::: "memory");
        Frag pf;
        pf.u[0] = *(const uint4*)(pl + ln * 32 + 16 * lh);
        pf.u[1] = *(const uint4*)(pl + ln * 32 + 16 * lh + 8);
        // P(16x32) @ V(32x64): V stored transposed -> contiguous B-frags
#pragma unroll
        for (int dt = 0; dt < 4; ++dt) {
            const _Float16* vrow =
                vbh + (size_t)(dt * 16 + ln) * SEQ + keybase + 16 * lh;
            Frag bv;
            bv.u[0] = *(const uint4*)(vrow);
            bv.u[1] = *(const uint4*)(vrow + 8);
            acc[dt] = __builtin_amdgcn_wmma_f32_16x16x32_f16(
                false, pf.h, false, bv.h, (short)0, acc[dt], false, false);
        }
    }

    // finalize: divide by row sums, write f16 (B, N, H*D) for projection GEMM
#pragma unroll
    for (int r = 0; r < 8; ++r) {
        const float inv = 1.0f / ssum[r];
        const int tok = qbase + r + 8 * lh;
        _Float16* orow = o16 + ((size_t)bb * SEQ + tok) * CDIM + h * HDIM;
#pragma unroll
        for (int dt = 0; dt < 4; ++dt)
            orow[dt * 16 + ln] = (_Float16)(acc[dt][r] * inv);
    }
}

// ----------------------------- projection GEMM ------------------------------
__global__ void __launch_bounds__(128)
attn_proj_gemm(const _Float16* __restrict__ a16, const _Float16* __restrict__ w16,
               const float* __restrict__ bias, float* __restrict__ out) {
    const int wave = (int)((blockIdx.x * blockDim.x + threadIdx.x) >> 5);
    const int lane = threadIdx.x & 31;
    const int NT = CDIM / 64;                 // 12 n-tiles
    const int mt = wave / NT, nt = wave % NT;
    const int mbase = mt * 32, nbase = nt * 64;
    const int ln = lane & 15, lh = lane >> 4;

    v8f acc[2][4];
    const v8f vzero = {0.f, 0.f, 0.f, 0.f, 0.f, 0.f, 0.f, 0.f};
#pragma unroll
    for (int i = 0; i < 2; ++i)
#pragma unroll
        for (int j = 0; j < 4; ++j) acc[i][j] = vzero;

    wmma_tile_32x64(a16, w16, CDIM, CDIM, CDIM, mbase, nbase, lane, acc);

#pragma unroll
    for (int t = 0; t < 4; ++t) {
        const int j = nbase + t * 16 + ln;
        const float bj = bias[j];
#pragma unroll
        for (int mi = 0; mi < 2; ++mi) {
#pragma unroll
            for (int r = 0; r < 8; ++r) {
                const int i = mbase + mi * 16 + r + 8 * lh;
                out[(size_t)i * CDIM + j] = acc[mi][t][r] + bj;
            }
        }
    }
}

// -------------------------------- launcher ----------------------------------
extern "C" void kernel_launch(void* const* d_in, const int* in_sizes, int n_in,
                              void* d_out, int out_size, void* d_ws, size_t ws_size,
                              hipStream_t stream) {
    const float* x      = (const float*)d_in[0];   // (16,1024,768)
    const float* mask   = (const float*)d_in[1];   // (16,1,1024,1024)
    const float* qkv_w  = (const float*)d_in[2];   // (2304,768)
    const float* qkv_b  = (const float*)d_in[3];   // (2304,)
    const float* proj_w = (const float*)d_in[4];   // (768,768)
    const float* proj_b = (const float*)d_in[5];   // (768,)
    float* out = (float*)d_out;

    char* ws = (char*)d_ws;
    size_t off = 0;
    auto carve = [&](size_t bytes) -> void* {
        void* p = ws + off;
        off += (bytes + 255) & ~(size_t)255;
        return p;
    };
    const size_t M = (size_t)BATCH * SEQ;          // 16384 tokens
    _Float16* x16    = (_Float16*)carve(M * CDIM * 2);
    _Float16* qkvw16 = (_Float16*)carve((size_t)QKVN * CDIM * 2);
    _Float16* pw16   = (_Float16*)carve((size_t)CDIM * CDIM * 2);
    _Float16* q16    = (_Float16*)carve((size_t)BATCH * HEADS * SEQ * HDIM * 2);
    _Float16* k16    = (_Float16*)carve((size_t)BATCH * HEADS * SEQ * HDIM * 2);
    _Float16* vT16   = (_Float16*)carve((size_t)BATCH * HEADS * HDIM * SEQ * 2);
    _Float16* ao16   = (_Float16*)carve(M * CDIM * 2);

    // 1) f32 -> f16 conversions
    const int nx = (int)(M * CDIM);
    const int nw = QKVN * CDIM;
    const int np = CDIM * CDIM;
    attn_cvt_f16<<<(nx + 255) / 256, 256, 0, stream>>>(x, x16, nx);
    attn_cvt_f16<<<(nw + 255) / 256, 256, 0, stream>>>(qkv_w, qkvw16, nw);
    attn_cvt_f16<<<(np + 255) / 256, 256, 0, stream>>>(proj_w, pw16, np);

    // 2) QKV GEMM: (16384 x 2304), 32x64 tile per wave, 4 waves per block
    {
        const int waves = (int)(M / 32) * (QKVN / 64);   // 512*36 = 18432
        attn_qkv_gemm<<<waves / 4, 128, 0, stream>>>(x16, qkvw16, qkv_b,
                                                     q16, k16, vT16);
    }
    // 3) flash attention: 16 query rows per wave
    {
        const int waves = BATCH * HEADS * (SEQ / 16);    // 12288
        attn_flash<<<waves / 4, 128, 0, stream>>>(q16, k16, vT16, mask, ao16);
    }
    // 4) projection GEMM: (16384 x 768) -> f32 output
    {
        const int waves = (int)(M / 32) * (CDIM / 64);   // 512*12 = 6144
        attn_proj_gemm<<<waves / 4, 128, 0, stream>>>(ao16, pw16, proj_b, out);
    }
}
